// MessagePassingStep_53137335386495
// MI455X (gfx1250) — matrix operand order repaired and checked by reading, not compiled, not run on hardware
//
#include <hip/hip_runtime.h>
#include <math.h>

#define NN 100000
#define NE 1000000
#define DIM 64
#define LN_EPS 1e-5f
#define STAGE_STRIDE 136   // 128 + 8 halfs padding (bank-conflict avoidance)

typedef _Float16 v16h __attribute__((ext_vector_type(16)));
typedef _Float16 v8h  __attribute__((ext_vector_type(8)));
typedef float    v8f  __attribute__((ext_vector_type(8)));

// Swizzled-weight layout in d_ws (units: halfs). Block = 512 halfs (32 lanes x 16).
#define EW1_OFF 0        // 192x128 : 6 ktiles x 8 ntiles = 48 blocks
#define EW2_OFF 24576    // 128x128 : 32 blocks
#define EW3_OFF 40960    // 128x64  : 16 blocks
#define NW1_OFF 49152    // 128x128 : 32 blocks
#define NW2_OFF 65536    // 128x128 : 32 blocks
#define NW3_OFF 81920    // 128x64  : 16 blocks

// Hardware tanh: CDNA5 lists TANH among transcendental VALU ops.
__device__ __forceinline__ float fast_tanh(float x) {
#if __has_builtin(__builtin_amdgcn_tanhf)
  return __builtin_amdgcn_tanhf(x);
#elif __has_builtin(__builtin_amdgcn_tanh_f32)
  return __builtin_amdgcn_tanh_f32(x);
#else
  float r;
  // v_nop covers the 1-op TRANS result-use hazard the compiler can't see.
  asm("v_tanh_f32 %0, %1\n\tv_nop" : "=v"(r) : "v"(x));
  return r;
#endif
}

// ---------------- setup kernels ----------------

__global__ void zero_f4(float4* __restrict__ p, int n4) {
  int i = blockIdx.x * blockDim.x + threadIdx.x;
  if (i < n4) p[i] = make_float4(0.f, 0.f, 0.f, 0.f);
}

// Pre-swizzle W[K][N] (f32, row-major) into per-lane WMMA B-fragment blocks (f16).
// blockHalf[lane*16 + j] = W[kt*32 + (lane>=16?8:0) + j + (j>=8?8:0)][nt*16 + (lane&15)]
__global__ void swizzle_w(const float* __restrict__ src, _Float16* __restrict__ dst,
                          int Nt, int Ncols, int totalHalfs) {
  int t = blockIdx.x * blockDim.x + threadIdx.x;
  if (t >= totalHalfs) return;
  int b = t >> 9, r = t & 511;
  int lane = r >> 4, j = r & 15;
  int kt = b / Nt, nt = b - kt * Nt;
  int k = kt * 32 + ((lane & 16) ? 8 : 0) + j + ((j & 8) ? 8 : 0);
  int n = nt * 16 + (lane & 15);
  dst[t] = (_Float16)src[(size_t)k * Ncols + n];
}

// ---------------- fragment helpers ----------------

// A-fragment (16x32 f16) from a global f32 row. lane<16: k in {0..7,16..23};
// lane>=16 (kb==8): k in {8..15,24..31}. rp points at the ktile start of the row.
__device__ __forceinline__ v16h load_a_global(const float* __restrict__ rp, int kb) {
  float4 f0 = *(const float4*)(rp + kb);
  float4 f1 = *(const float4*)(rp + kb + 4);
  float4 f2 = *(const float4*)(rp + kb + 16);
  float4 f3 = *(const float4*)(rp + kb + 20);
  v16h a;
  a[0]=(_Float16)f0.x; a[1]=(_Float16)f0.y; a[2]=(_Float16)f0.z; a[3]=(_Float16)f0.w;
  a[4]=(_Float16)f1.x; a[5]=(_Float16)f1.y; a[6]=(_Float16)f1.z; a[7]=(_Float16)f1.w;
  a[8]=(_Float16)f2.x; a[9]=(_Float16)f2.y; a[10]=(_Float16)f2.z; a[11]=(_Float16)f2.w;
  a[12]=(_Float16)f3.x; a[13]=(_Float16)f3.y; a[14]=(_Float16)f3.z; a[15]=(_Float16)f3.w;
  return a;
}

// aLane = stage + laneN*STAGE_STRIDE + kb (per-lane base; offsets are constants)
__device__ __forceinline__ v16h load_a_lds(const _Float16* aLane, int ktOff) {
  v8h lo = *(const v8h*)(aLane + ktOff);
  v8h hi = *(const v8h*)(aLane + ktOff + 16);
  v16h a;
#pragma unroll
  for (int i = 0; i < 8; ++i) { a[i] = lo[i]; a[8 + i] = hi[i]; }
  return a;
}

// wLane = layer_base + lane*16 (per-lane base; block offsets are constants)
__device__ __forceinline__ v16h load_b_lds(const _Float16* wLane, int blkOff) {
  v8h lo = *(const v8h*)(wLane + blkOff);
  v8h hi = *(const v8h*)(wLane + blkOff + 8);
  v16h b;
#pragma unroll
  for (int i = 0; i < 8; ++i) { b[i] = lo[i]; b[8 + i] = hi[i]; }
  return b;
}

// acc[NT] += stage(16 x KT*32) @ W(KT*32 x NT*16); weights pre-swizzled in LDS.
template <int NT, int KT>
__device__ __forceinline__ void matmul_stage(const _Float16* stage, const _Float16* wbase,
                                             int lane, v8f* acc) {
  const int laneN = lane & 15;
  const int kb = (lane & 16) ? 8 : 0;
  const _Float16* aLane = stage + laneN * STAGE_STRIDE + kb;  // one add, const offsets
  const _Float16* wLane = wbase + lane * 16;                  // one add, const offsets
#pragma unroll
  for (int kt = 0; kt < KT; ++kt) {
    v16h a = load_a_lds(aLane, kt * 32);
#pragma unroll
    for (int nt = 0; nt < NT; ++nt) {
      v16h b = load_b_lds(wLane, (kt * NT + nt) * 512);
      acc[nt] = __builtin_amdgcn_wmma_f32_16x16x32_f16(false, a, false, b,
                                                       (short)0, acc[nt], false, false);
    }
  }
}

// D-layout (lane<16: rows 0..7; lane>=16: rows 8..15; col = nt*16 + lane&15)
// -> bias + hw-tanh -> f16 staging buffer [16][STAGE_STRIDE].
template <int NT>
__device__ __forceinline__ void bias_tanh_stage(_Float16* stage, const float* __restrict__ bias,
                                                int lane, const v8f* acc) {
  const int laneN = lane & 15;
  const int mb = (lane & 16) ? 8 : 0;
  _Float16* sLane = stage + mb * STAGE_STRIDE + laneN;        // one add, const offsets
#pragma unroll
  for (int nt = 0; nt < NT; ++nt) {
    float bv = bias[nt * 16 + laneN];
#pragma unroll
    for (int v = 0; v < 8; ++v)
      sLane[v * STAGE_STRIDE + nt * 16] = (_Float16)fast_tanh(acc[nt][v] + bv);
  }
}

// Half-wave (16-lane) LayerNorm stats from 4 D-accumulators covering 64 cols.
__device__ __forceinline__ void ln_stats(const v8f* acc3, float* mu, float* rstd) {
#pragma unroll
  for (int v = 0; v < 8; ++v) {
    float s = acc3[0][v] + acc3[1][v] + acc3[2][v] + acc3[3][v];
    float q = acc3[0][v]*acc3[0][v] + acc3[1][v]*acc3[1][v] +
              acc3[2][v]*acc3[2][v] + acc3[3][v]*acc3[3][v];
#pragma unroll
    for (int off = 8; off; off >>= 1) {
      s += __shfl_xor(s, off, 32);
      q += __shfl_xor(q, off, 32);
    }
    float m = s * (1.0f / 64.0f);
    mu[v] = m;
    rstd[v] = rsqrtf(q * (1.0f / 64.0f) - m * m + LN_EPS);
  }
}

// ---------------- edge MLP + antisymmetric scatter ----------------

__global__ __launch_bounds__(256) void edge_mlp(
    const float* __restrict__ x, const int* __restrict__ senders,
    const int* __restrict__ receivers, const float* __restrict__ edge_attr,
    const _Float16* __restrict__ wsw,
    const float* __restrict__ b1, const float* __restrict__ b2,
    const float* __restrict__ b3, const float* __restrict__ gamma,
    const float* __restrict__ beta,
    float* __restrict__ out_edge, float* __restrict__ agg, int nTiles) {
  __shared__ __align__(64) _Float16 sW[49152];                 // 96 KB weights
  __shared__ __align__(64) _Float16 sStage[8][16 * STAGE_STRIDE];
  {
    const uint4* src = (const uint4*)wsw;   // EW1..EW3 contiguous at offset 0
    uint4* dst = (uint4*)sW;
    for (int i = threadIdx.x; i < 6144; i += 256) dst[i] = src[i];
  }
  __syncthreads();

  const int lane = threadIdx.x & 31;
  const int wave = threadIdx.x >> 5;
  const int laneN = lane & 15;
  const int kb = (lane & 16) ? 8 : 0;
  const int mb = kb;  // same 0/8 split for D rows
  _Float16* stage = sStage[wave];
  const int stride = gridDim.x * 8;
  const v8f vzero = {0.f,0.f,0.f,0.f,0.f,0.f,0.f,0.f};

  for (int tile = blockIdx.x * 8 + wave; tile < nTiles; tile += stride) {
    const int e0 = tile * 16;
    const int eA = e0 + laneN;
    const int rrow = receivers[eA];
    const int srow = senders[eA];

    // layer 1: [16x192] @ [192x128]
    v8f acc[8];
#pragma unroll
    for (int nt = 0; nt < 8; ++nt) acc[nt] = vzero;
    {
      const _Float16* wLane = sW + lane * 16;   // EW1 base + lane
#pragma unroll
      for (int kt = 0; kt < 6; ++kt) {
        const float* rp = (kt < 2) ? edge_attr + (size_t)eA * DIM + kt * 32
                        : (kt < 4) ? x + (size_t)rrow * DIM + (kt - 2) * 32
                                   : x + (size_t)srow * DIM + (kt - 4) * 32;
        v16h a = load_a_global(rp, kb);
#pragma unroll
        for (int nt = 0; nt < 8; ++nt) {
          v16h b = load_b_lds(wLane, (kt * 8 + nt) * 512);
          acc[nt] = __builtin_amdgcn_wmma_f32_16x16x32_f16(false, a, false, b,
                                                           (short)0, acc[nt], false, false);
        }
      }
    }
    bias_tanh_stage<8>(stage, b1, lane, acc);

    // layer 2: [16x128] @ [128x128]
#pragma unroll
    for (int nt = 0; nt < 8; ++nt) acc[nt] = vzero;
    matmul_stage<8, 4>(stage, sW + EW2_OFF, lane, acc);
    bias_tanh_stage<8>(stage, b2, lane, acc);

    // layer 3: [16x128] @ [128x64]
    v8f acc3[4];
#pragma unroll
    for (int nt = 0; nt < 4; ++nt) acc3[nt] = vzero;
    matmul_stage<4, 4>(stage, sW + EW3_OFF, lane, acc3);
#pragma unroll
    for (int nt = 0; nt < 4; ++nt) {
      float bv = b3[nt * 16 + laneN];
#pragma unroll
      for (int v = 0; v < 8; ++v) acc3[nt][v] += bv;
    }

    float mu[8], rstd[8];
    ln_stats(acc3, mu, rstd);

    int rid[8], sid[8];
#pragma unroll
    for (int v = 0; v < 8; ++v) {
      rid[v] = receivers[e0 + mb + v];
      sid[v] = senders[e0 + mb + v];
    }
#pragma unroll
    for (int nt = 0; nt < 4; ++nt) {
      int c = nt * 16 + laneN;
      float g = gamma[c], bt = beta[c];
#pragma unroll
      for (int v = 0; v < 8; ++v) {
        float msg = (acc3[nt][v] - mu[v]) * rstd[v] * g + bt;
        size_t em = (size_t)(e0 + mb + v);
        out_edge[em * DIM + c] = edge_attr[em * DIM + c] + msg;
        unsafeAtomicAdd(&agg[(size_t)rid[v] * DIM + c], msg);
        unsafeAtomicAdd(&agg[(size_t)sid[v] * DIM + c], -msg);
      }
    }
  }
}

// ---------------- node MLP + residual ----------------

__global__ __launch_bounds__(256) void node_mlp(
    const float* __restrict__ x, const float* __restrict__ agg,
    const _Float16* __restrict__ wsw,
    const float* __restrict__ b1, const float* __restrict__ b2,
    const float* __restrict__ b3, const float* __restrict__ gamma,
    const float* __restrict__ beta,
    float* __restrict__ out_node, int nTiles) {
  __shared__ __align__(64) _Float16 sW[40960];                 // 80 KB weights
  __shared__ __align__(64) _Float16 sStage[8][16 * STAGE_STRIDE];
  {
    const uint4* src = (const uint4*)wsw + 6144;  // NW1 at half-offset 49152
    uint4* dst = (uint4*)sW;
    for (int i = threadIdx.x; i < 5120; i += 256) dst[i] = src[i];
  }
  __syncthreads();

  const int lane = threadIdx.x & 31;
  const int wave = threadIdx.x >> 5;
  const int laneN = lane & 15;
  const int kb = (lane & 16) ? 8 : 0;
  const int mb = kb;
  _Float16* stage = sStage[wave];
  const int stride = gridDim.x * 8;
  const v8f vzero = {0.f,0.f,0.f,0.f,0.f,0.f,0.f,0.f};

  for (int tile = blockIdx.x * 8 + wave; tile < nTiles; tile += stride) {
    const int n0 = tile * 16;
    const int nA = n0 + laneN;

    // layer 1: [16x128] @ [128x128], cols 0..63 = x, 64..127 = agg
    v8f acc[8];
#pragma unroll
    for (int nt = 0; nt < 8; ++nt) acc[nt] = vzero;
    {
      const _Float16* wLane = sW + lane * 16;   // NW1 base + lane
#pragma unroll
      for (int kt = 0; kt < 4; ++kt) {
        const float* rp = (kt < 2) ? x + (size_t)nA * DIM + kt * 32
                                   : agg + (size_t)nA * DIM + (kt - 2) * 32;
        v16h a = load_a_global(rp, kb);
#pragma unroll
        for (int nt = 0; nt < 8; ++nt) {
          v16h b = load_b_lds(wLane, (kt * 8 + nt) * 512);
          acc[nt] = __builtin_amdgcn_wmma_f32_16x16x32_f16(false, a, false, b,
                                                           (short)0, acc[nt], false, false);
        }
      }
    }
    bias_tanh_stage<8>(stage, b1, lane, acc);

    // layer 2
#pragma unroll
    for (int nt = 0; nt < 8; ++nt) acc[nt] = vzero;
    matmul_stage<8, 4>(stage, sW + (NW2_OFF - NW1_OFF), lane, acc);
    bias_tanh_stage<8>(stage, b2, lane, acc);

    // layer 3
    v8f acc3[4];
#pragma unroll
    for (int nt = 0; nt < 4; ++nt) acc3[nt] = vzero;
    matmul_stage<4, 4>(stage, sW + (NW3_OFF - NW1_OFF), lane, acc3);
#pragma unroll
    for (int nt = 0; nt < 4; ++nt) {
      float bv = b3[nt * 16 + laneN];
#pragma unroll
      for (int v = 0; v < 8; ++v) acc3[nt][v] += bv;
    }

    float mu[8], rstd[8];
    ln_stats(acc3, mu, rstd);

#pragma unroll
    for (int nt = 0; nt < 4; ++nt) {
      int c = nt * 16 + laneN;
      float g = gamma[c], bt = beta[c];
#pragma unroll
      for (int v = 0; v < 8; ++v) {
        float gx = (acc3[nt][v] - mu[v]) * rstd[v] * g + bt;
        size_t nn = (size_t)(n0 + mb + v);
        out_node[nn * DIM + c] = x[nn * DIM + c] + gx;  // residual (overwrites agg)
      }
    }
  }
}

// ---------------- launch ----------------

extern "C" void kernel_launch(void* const* d_in, const int* in_sizes, int n_in,
                              void* d_out, int out_size, void* d_ws, size_t ws_size,
                              hipStream_t stream) {
  (void)in_sizes; (void)n_in; (void)out_size; (void)ws_size;
  const float* x         = (const float*)d_in[0];
  const int*   senders   = (const int*)d_in[1];
  const int*   receivers = (const int*)d_in[2];
  const float* edge_attr = (const float*)d_in[3];
  const float* ew1 = (const float*)d_in[4];
  const float* eb1 = (const float*)d_in[5];
  const float* ew2 = (const float*)d_in[6];
  const float* eb2 = (const float*)d_in[7];
  const float* ew3 = (const float*)d_in[8];
  const float* eb3 = (const float*)d_in[9];
  const float* eg  = (const float*)d_in[10];
  const float* ebt = (const float*)d_in[11];
  const float* nw1 = (const float*)d_in[12];
  const float* nb1 = (const float*)d_in[13];
  const float* nw2 = (const float*)d_in[14];
  const float* nb2 = (const float*)d_in[15];
  const float* nw3 = (const float*)d_in[16];
  const float* nb3 = (const float*)d_in[17];
  const float* ng  = (const float*)d_in[18];
  const float* nbt = (const float*)d_in[19];

  float* out      = (float*)d_out;
  float* agg      = out;                       // node-out region doubles as f32 accumulator
  float* out_edge = out + (size_t)NN * DIM;
  _Float16* wsw   = (_Float16*)d_ws;           // 176 KB of swizzled f16 weights

  // zero the aggregation buffer (N*64 floats = 1.6M float4, exact)
  zero_f4<<<6250, 256, 0, stream>>>((float4*)agg, NN * DIM / 4);

  // pre-swizzle weights into WMMA B-fragment blocks
  swizzle_w<<<96, 256, 0, stream>>>(ew1, wsw + EW1_OFF, 8, 128, 24576);
  swizzle_w<<<64, 256, 0, stream>>>(ew2, wsw + EW2_OFF, 8, 128, 16384);
  swizzle_w<<<32, 256, 0, stream>>>(ew3, wsw + EW3_OFF, 4,  64,  8192);
  swizzle_w<<<64, 256, 0, stream>>>(nw1, wsw + NW1_OFF, 8, 128, 16384);
  swizzle_w<<<64, 256, 0, stream>>>(nw2, wsw + NW2_OFF, 8, 128, 16384);
  swizzle_w<<<32, 256, 0, stream>>>(nw3, wsw + NW3_OFF, 4,  64,  8192);

  edge_mlp<<<1024, 256, 0, stream>>>(x, senders, receivers, edge_attr, wsw,
                                     eb1, eb2, eb3, eg, ebt,
                                     out_edge, agg, NE / 16);
  node_mlp<<<512, 256, 0, stream>>>(x, agg, wsw, nb1, nb2, nb3, ng, nbt,
                                    out, NN / 16);
}
